// MultiHeadAttention_18219251269960
// MI455X (gfx1250) — matrix operand order repaired and checked by reading, not compiled, not run on hardware
//
#include <hip/hip_runtime.h>
#include <hip/hip_bf16.h>

// Problem constants (match reference)
#define BATCH   4
#define SEQ     2048
#define DMODEL  1024
#define NHEAD   16
#define DK      64
#define NEGVAL  (-1e9f)

typedef __attribute__((ext_vector_type(16))) _Float16     v16h;
typedef __attribute__((ext_vector_type(8)))  float        v8f;
typedef __attribute__((ext_vector_type(4)))  unsigned int v4u;
typedef __attribute__((ext_vector_type(4)))  int          v4i;
typedef __attribute__((ext_vector_type(8)))  int          v8i;

__device__ __forceinline__ v8f wmma_f16(v16h a, v16h b, v8f c) {
  return __builtin_amdgcn_wmma_f32_16x16x32_f16(
      false, a, false, b, (short)0, c, false, false);
}

// ---------------------------------------------------------------------------
// TDM: async 2D tile load Global -> LDS (CDNA5 ISA ch.8 D# descriptor).
//   group0: [1:0]=count=1, [63:32]=lds_addr, [120:64]=global_addr, [127:126]=type=2
//   group1: [17:16]=data_size, [79:48]=tensor_dim0, [111:80]=tensor_dim1,
//           [127:112]=tile_dim0, [143:128]=tile_dim1, [207:160]=dim0_stride
// All dims/strides in data_size units. Tracked by TENSORcnt.
// ---------------------------------------------------------------------------
__device__ __forceinline__ void tdm_load_2d(unsigned ldsOff, const void* gptr,
                                            unsigned dsizeCode,  // 1=2B, 2=4B
                                            unsigned tensorDim0, unsigned tensorDim1,
                                            unsigned stride0,
                                            unsigned tileDim0, unsigned tileDim1) {
  unsigned long long ga = (unsigned long long)(__UINTPTR_TYPE__)gptr;
  v4u g0;
  g0[0] = 1u;                                             // count=1 (valid)
  g0[1] = ldsOff;                                         // lds_addr (bytes)
  g0[2] = (unsigned)(ga & 0xFFFFFFFFu);                   // global_addr[31:0]
  g0[3] = (unsigned)((ga >> 32) & 0x01FFFFFFu) | (2u << 30);  // [56:32] | type=2
  v8i g1;
  g1[0] = (int)(dsizeCode << 16);                         // data_size
  g1[1] = (int)((tensorDim0 & 0xFFFFu) << 16);            // tensor_dim0[15:0]
  g1[2] = (int)(((tensorDim0 >> 16) & 0xFFFFu) |          // tensor_dim0[31:16]
                ((tensorDim1 & 0xFFFFu) << 16));          // tensor_dim1[15:0]
  g1[3] = (int)(((tensorDim1 >> 16) & 0xFFFFu) |          // tensor_dim1[31:16]
                ((tileDim0 & 0xFFFFu) << 16));            // tile_dim0
  g1[4] = (int)(tileDim1 & 0xFFFFu);                      // tile_dim1 (tile_dim2=0)
  g1[5] = (int)stride0;                                   // dim0_stride[31:0]
  g1[6] = 0;                                              // stride hi / dim1_stride
  g1[7] = 0;
  v4i g2 = {0, 0, 0, 0};
  v4i g3 = {0, 0, 0, 0};
#if defined(__clang_major__) && __clang_major__ >= 23
  v8i g4 = {0, 0, 0, 0, 0, 0, 0, 0};
  __builtin_amdgcn_tensor_load_to_lds(g0, g1, g2, g3, g4, 0);
#else
  __builtin_amdgcn_tensor_load_to_lds(g0, g1, g2, g3, 0);
#endif
}

__device__ __forceinline__ unsigned lds_offset_of(const void* p) {
  // LDS aperture keeps the workgroup-relative offset in addr[31:0] (ISA 10.2)
  return (unsigned)(unsigned long long)(__UINTPTR_TYPE__)p;
}

// ---------------------------------------------------------------------------
// C[M,N] = A[M,K] * W[N,K]^T, TDM-staged, LDS double-buffered.
// Block = 256 thr (8 waves) computes a 128x128 C tile; wave w does rows
// w*16..w*16+15 over all 128 cols: 1 A-fragment reused by 8 B-fragments
// -> 8 WMMAs per 32-wide K step. Wave 0 drives the TDM pipeline.
// ---------------------------------------------------------------------------
template<typename AT, typename OT>
__global__ void __launch_bounds__(256)
gemm_xwT(const AT* __restrict__ A, const float* __restrict__ W,
         OT* __restrict__ C, int M, int N, int K) {
  __shared__ __align__(16) unsigned char sAraw[2][128 * 32 * 4];
  __shared__ __align__(16) float sW[2][128 * 32];

  const int w    = (int)(threadIdx.x >> 5);
  const int lane = (int)(threadIdx.x & 31);
  const int nblk = N >> 7;                       // 128-col tiles
  const int mtB  = ((int)blockIdx.x / nblk) * 128;
  const int ntB  = ((int)blockIdx.x - ((int)blockIdx.x / nblk) * nblk) * 128;

  const int n16   = lane & 15;
  const int aBase = (lane < 16) ? 0 : 8;
  const int bBase = (lane < 16) ? 0 : 8;
  const unsigned dsz = (sizeof(AT) == 4) ? 2u : 1u;

  // Prologue: fill buffer 0
  if (threadIdx.x < 32) {
    tdm_load_2d(lds_offset_of(&sAraw[0][0]), A + (size_t)mtB * K,
                dsz, (unsigned)K, (unsigned)M, (unsigned)K, 32u, 128u);
    tdm_load_2d(lds_offset_of(&sW[0][0]),    W + (size_t)ntB * K,
                2u,  (unsigned)K, (unsigned)N, (unsigned)K, 32u, 128u);
  }

  v8f acc[8] = {v8f{}, v8f{}, v8f{}, v8f{}, v8f{}, v8f{}, v8f{}, v8f{}};

  for (int kk = 0; kk < K; kk += 32) {
    const int buf = (kk >> 5) & 1;
    if (threadIdx.x < 32) {
      if (kk + 32 < K) {
        tdm_load_2d(lds_offset_of(&sAraw[buf ^ 1][0]),
                    A + (size_t)mtB * K + (kk + 32),
                    dsz, (unsigned)K, (unsigned)M, (unsigned)K, 32u, 128u);
        tdm_load_2d(lds_offset_of(&sW[buf ^ 1][0]),
                    W + (size_t)ntB * K + (kk + 32),
                    2u,  (unsigned)K, (unsigned)N, (unsigned)K, 32u, 128u);
        __builtin_amdgcn_s_wait_tensorcnt((short)2);  // current buffer ready
      } else {
        __builtin_amdgcn_s_wait_tensorcnt((short)0);
      }
    }
    __syncthreads();                                  // publish staged tiles

    const AT*    sAb = (const AT*)&sAraw[buf][0];
    const float* sWb = &sW[buf][0];

    v16h af;
#pragma unroll
    for (int j = 0; j < 8; ++j) {
      af[j]     = (_Float16)sAb[(w * 16 + n16) * 32 + aBase + j];
      af[j + 8] = (_Float16)sAb[(w * 16 + n16) * 32 + 16 + aBase + j];
    }
#pragma unroll
    for (int nsub = 0; nsub < 8; ++nsub) {
      v16h bf;
#pragma unroll
      for (int j = 0; j < 16; ++j)
        bf[j] = (_Float16)sWb[(nsub * 16 + n16) * 32 + bBase + j + (j & 8)];
      acc[nsub] = wmma_f16(af, bf, acc[nsub]);
    }
    __syncthreads();    // all reads done before buffer is overwritten
  }

  const int rOff = (lane < 16) ? 0 : 8;
#pragma unroll
  for (int nsub = 0; nsub < 8; ++nsub) {
#pragma unroll
    for (int r = 0; r < 8; ++r)
      C[(size_t)(mtB + w * 16 + rOff + r) * N + ntB + nsub * 16 + n16]
          = (OT)acc[nsub][r];
  }
}

// ---------------------------------------------------------------------------
// Flash attention: O = softmax(Q K^T / sqrt(DK) [+mask]) V  per (b,h).
// Q/K/V f16 as [B*S, DMODEL], head h in columns h*DK..h*DK+63.
// One wave per (b, h, 16-row q tile); 32-key blocks; online softmax;
// P transposed C-layout -> A-layout through LDS.
// ---------------------------------------------------------------------------
__global__ void __launch_bounds__(256)
attn_kernel(const _Float16* __restrict__ Q, const _Float16* __restrict__ Kp,
            const _Float16* __restrict__ Vp, const int* __restrict__ mask,
            _Float16* __restrict__ O) {
  const int waveInBlk = (int)(threadIdx.x >> 5);
  const int wave = (int)blockIdx.x * (int)(blockDim.x >> 5) + waveInBlk;
  const int lane = (int)(threadIdx.x & 31);

  const int qtiles = SEQ / 16;
  const int qt = wave % qtiles;
  const int bh = wave / qtiles;
  const int h  = bh % NHEAD;
  const int b  = bh / NHEAD;

  const int n16   = lane & 15;
  const int aBase = (lane < 16) ? 0 : 8;
  const int bBase = (lane < 16) ? 0 : 8;
  const int rOff  = aBase;

  __shared__ _Float16 psh[8][16 * 32];
  _Float16* pb = &psh[waveInBlk][0];

  const size_t qRow = (size_t)(b * SEQ + qt * 16 + n16) * DMODEL + h * DK;
  v16h qa0, qa1;
#pragma unroll
  for (int j = 0; j < 8; ++j) {
    qa0[j]     = Q[qRow + aBase + j];
    qa0[j + 8] = Q[qRow + 16 + aBase + j];
    qa1[j]     = Q[qRow + 32 + aBase + j];
    qa1[j + 8] = Q[qRow + 48 + aBase + j];
  }

  v8f oacc[4] = {v8f{}, v8f{}, v8f{}, v8f{}};
  float mrun[8], lrun[8];
#pragma unroll
  for (int r = 0; r < 8; ++r) { mrun[r] = -1e30f; lrun[r] = 0.0f; }

  const float scale = 0.125f;
  const int qrowBase = qt * 16 + rOff;

  for (int kb = 0; kb < SEQ; kb += 32) {
    const size_t kRowL = (size_t)(b * SEQ + kb + n16) * DMODEL + h * DK;
    const size_t kRowR = kRowL + (size_t)16 * DMODEL;
    v16h kL0, kL1, kR0, kR1;
#pragma unroll
    for (int j = 0; j < 16; ++j) {
      kL0[j] = Kp[kRowL + bBase + j + (j & 8)];
      kL1[j] = Kp[kRowL + 32 + bBase + j + (j & 8)];
      kR0[j] = Kp[kRowR + bBase + j + (j & 8)];
      kR1[j] = Kp[kRowR + 32 + bBase + j + (j & 8)];
    }
    v8f sL = {}, sR = {};
    sL = wmma_f16(qa0, kL0, sL);
    sL = wmma_f16(qa1, kL1, sL);
    sR = wmma_f16(qa0, kR0, sR);
    sR = wmma_f16(qa1, kR1, sR);

#pragma unroll
    for (int r = 0; r < 8; ++r) {
      float vL = sL[r] * scale;
      float vR = sR[r] * scale;
      const int qr = qrowBase + r;
      if (mask[(size_t)qr * SEQ + kb + n16] == 0)      vL = NEGVAL;
      if (mask[(size_t)qr * SEQ + kb + 16 + n16] == 0) vR = NEGVAL;

      float tm = fmaxf(vL, vR);
#pragma unroll
      for (int o = 1; o < 16; o <<= 1) tm = fmaxf(tm, __shfl_xor(tm, o, 32));
      const float mnew = fmaxf(mrun[r], tm);
      const float pL = __expf(vL - mnew);
      const float pR = __expf(vR - mnew);
      float rs = pL + pR;
#pragma unroll
      for (int o = 1; o < 16; o <<= 1) rs += __shfl_xor(rs, o, 32);
      const float corr = __expf(mrun[r] - mnew);
      lrun[r] = lrun[r] * corr + rs;
      mrun[r] = mnew;
      oacc[0][r] *= corr; oacc[1][r] *= corr;
      oacc[2][r] *= corr; oacc[3][r] *= corr;

      pb[(rOff + r) * 32 + n16]      = (_Float16)pL;
      pb[(rOff + r) * 32 + 16 + n16] = (_Float16)pR;
    }
    __syncthreads();

    v16h pa;
#pragma unroll
    for (int j = 0; j < 8; ++j) {
      pa[j]     = pb[n16 * 32 + aBase + j];
      pa[j + 8] = pb[n16 * 32 + 16 + aBase + j];
    }
#pragma unroll
    for (int f = 0; f < 4; ++f) {
      v16h vf;
#pragma unroll
      for (int j = 0; j < 16; ++j)
        vf[j] = Vp[(size_t)(b * SEQ + kb + bBase + j + (j & 8)) * DMODEL
                   + h * DK + f * 16 + n16];
      oacc[f] = wmma_f16(pa, vf, oacc[f]);
    }
    __syncthreads();
  }

#pragma unroll
  for (int f = 0; f < 4; ++f) {
#pragma unroll
    for (int r = 0; r < 8; ++r) {
      O[(size_t)(b * SEQ + qt * 16 + rOff + r) * DMODEL + h * DK + f * 16 + n16]
          = (_Float16)(oacc[f][r] / lrun[r]);
    }
  }
}

// ---------------------------------------------------------------------------
extern "C" void kernel_launch(void* const* d_in, const int* in_sizes, int n_in,
                              void* d_out, int out_size, void* d_ws, size_t ws_size,
                              hipStream_t stream) {
  const float* q    = (const float*)d_in[0];
  const float* k    = (const float*)d_in[1];
  const float* v    = (const float*)d_in[2];
  const int*   mask = (const int*)  d_in[3];
  const float* w_q  = (const float*)d_in[4];
  const float* w_k  = (const float*)d_in[5];
  const float* w_v  = (const float*)d_in[6];
  const float* w_o  = (const float*)d_in[7];
  float* out = (float*)d_out;

  constexpr int M = BATCH * SEQ;     // 8192
  constexpr int N = DMODEL;          // 1024
  constexpr int K = DMODEL;          // 1024
  constexpr size_t planeElems = (size_t)M * DMODEL;

  _Float16* Qws = (_Float16*)d_ws;
  _Float16* Kws = Qws + planeElems;
  _Float16* Vws = Kws + planeElems;
  _Float16* Ows = Vws + planeElems;

  dim3 blk(256);
  dim3 grdGemm((M / 128) * (N / 128));                 // 512 blocks, 128x128 tiles
  dim3 grdAttn(BATCH * NHEAD * (SEQ / 16) / 8);        // 1024 blocks

  gemm_xwT<float, _Float16><<<grdGemm, blk, 0, stream>>>(q, w_q, Qws, M, N, K);
  gemm_xwT<float, _Float16><<<grdGemm, blk, 0, stream>>>(k, w_k, Kws, M, N, K);
  gemm_xwT<float, _Float16><<<grdGemm, blk, 0, stream>>>(v, w_v, Vws, M, N, K);

  attn_kernel<<<grdAttn, blk, 0, stream>>>(Qws, Kws, Vws, mask, Ows);

  gemm_xwT<_Float16, float><<<grdGemm, blk, 0, stream>>>(Ows, w_o, out, M, N, K);
}
